// MultiHeadAttention_24292335026463
// MI455X (gfx1250) — matrix operand order repaired and checked by reading, not compiled, not run on hardware
//
#include <hip/hip_runtime.h>
#include <hip/hip_bf16.h>

// ---------------------------------------------------------------------------
// MultiHeadAttention forward for MI455X (gfx1250, wave32, WMMA f16->f32)
// B=4, S=1024, D=1024, H=16, DK=64
// Outputs (concatenated in d_out): att [B,H,S,S] f32, out [B,S,D] f32
//
// Pipeline:
//   proj<0> q,k -> f16 [B,H,S,DK]; proj<1> v -> f16 [B,H,DK,S]
//   fused_attn: scores (WMMA) -> softmax (LDS-resident) -> P@V (WMMA)
//               writes att to HBM exactly once, never re-reads it
//   proj<2>: ctx @ Wo^T + bo -> f32 out
// ---------------------------------------------------------------------------

typedef __attribute__((ext_vector_type(16))) _Float16 v16h;
typedef __attribute__((ext_vector_type(8)))  _Float16 v8h;
typedef __attribute__((ext_vector_type(8)))  float    v8f;
typedef __attribute__((ext_vector_type(4)))  float    v4f;

constexpr int BATCH = 4;
constexpr int SEQ   = 1024;
constexpr int DIM   = 1024;
constexpr int NH    = 16;
constexpr int DK    = 64;
constexpr int LDH   = 40;    // proj kernels: LDS row stride in halves
constexpr int TLD   = 72;    // fused kernel: Q/K/V tile row stride in halves
constexpr int SC_LD = 1036;  // fused kernel: score row stride in floats
                             // (12 mod 64 banks/row -> conflict-free frag reads)

// ---------------- WMMA helpers --------------------------------------------

__device__ inline v8f wmma_f16(v16h a, v16h b, v8f c) {
  return __builtin_amdgcn_wmma_f32_16x16x32_f16(
      /*neg_a=*/false, a, /*neg_b=*/false, b,
      /*c_mod=*/(short)0, c, /*reuse_a=*/false, /*reuse_b=*/false);
}

// A fragment: 16x32 f16 tile. Lane L: m=L&15, half=L>>4;
// elems [0..7] = K half*8.., [8..15] = K 16+half*8..  (per ISA A-layout table)
template <int LD>
__device__ inline v16h frag_a_h(const _Float16* sm, int row0, int k0, int lane) {
  const int m  = row0 + (lane & 15);
  const int h2 = lane >> 4;
  const _Float16* p = sm + m * LD + k0;
  v8h lo = *(const v8h*)(p + h2 * 8);
  v8h hi = *(const v8h*)(p + 16 + h2 * 8);
  v16h r;
#pragma unroll
  for (int i = 0; i < 8; ++i) { r[i] = lo[i]; r[i + 8] = hi[i]; }
  return r;
}

// B fragment: 32x16 (KxN) tile, LDS stored N-major sm[n*LD + k].
// Lane L: n=L&15, half=L>>4; elems [0..7] = K half*8.., [8..15] = K 16+half*8..
template <int LD>
__device__ inline v16h frag_b_h(const _Float16* sm, int n0, int k0, int lane) {
  const int n  = n0 + (lane & 15);
  const int h2 = lane >> 4;
  const _Float16* p = sm + n * LD + k0;
  v8h lo = *(const v8h*)(p + h2 * 8);
  v8h hi = *(const v8h*)(p + 16 + h2 * 8);
  v16h r;
#pragma unroll
  for (int i = 0; i < 8; ++i) { r[i] = lo[i]; r[i + 8] = hi[i]; }
  return r;
}

// A fragment from f32 LDS (probabilities), converted to f16 in registers.
__device__ inline v16h frag_a_f32(const float* sm, int row0, int col0, int lane) {
  const int m  = row0 + (lane & 15);
  const int h2 = lane >> 4;
  const float* p = sm + (size_t)m * SC_LD + col0;
  v4f f0 = *(const v4f*)(p + h2 * 8);
  v4f f1 = *(const v4f*)(p + h2 * 8 + 4);
  v4f f2 = *(const v4f*)(p + 16 + h2 * 8);
  v4f f3 = *(const v4f*)(p + 16 + h2 * 8 + 4);
  v16h r;
#pragma unroll
  for (int i = 0; i < 4; ++i) {
    r[i]      = (_Float16)f0[i];
    r[i + 4]  = (_Float16)f1[i];
    r[i + 8]  = (_Float16)f2[i];
    r[i + 12] = (_Float16)f3[i];
  }
  return r;
}

// ---------------- LDS tile loader (proj kernels, 256 threads) --------------

// 128 rows x 32 cols, f32 global -> f16 LDS. 2 threads/row, 16 f32 each.
__device__ inline void load_tile_f32(_Float16* s, const float* g, int row0,
                                     size_t ldg, int k0, int tid) {
  const int r = tid >> 1;
  const int c = (tid & 1) * 16;
  const float* gp = g + (size_t)(row0 + r) * ldg + k0 + c;
  v4f f0 = *(const v4f*)(gp + 0);
  v4f f1 = *(const v4f*)(gp + 4);
  v4f f2 = *(const v4f*)(gp + 8);
  v4f f3 = *(const v4f*)(gp + 12);
  v8h h0, h1;
#pragma unroll
  for (int i = 0; i < 4; ++i) {
    h0[i]     = (_Float16)f0[i];
    h0[i + 4] = (_Float16)f1[i];
    h1[i]     = (_Float16)f2[i];
    h1[i + 4] = (_Float16)f3[i];
  }
  _Float16* sp = s + r * LDH + c;
  *(v8h*)(sp)     = h0;
  *(v8h*)(sp + 8) = h1;
}

// ---------------- Projection GEMM: Out = X @ W^T + bias --------------------
// MODE 0: f16 out [B,H,S,DK]   (q, k)
// MODE 1: f16 out [B,H,DK,S]   (v, pre-transposed for P@V)
// MODE 2: f32 out [B,S,D]      (final projection)
template <int MODE>
__global__ void __launch_bounds__(256)
proj_kernel(const float* __restrict__ X, const float* __restrict__ W,
            const float* __restrict__ bias, void* __restrict__ out) {
  __shared__ _Float16 As[128 * LDH];
  __shared__ _Float16 Bs[128 * LDH];

  const int tid  = threadIdx.x;
  const int lane = tid & 31;
  const int wid  = tid >> 5;
  const int m0   = blockIdx.y * 128;
  const int n0   = blockIdx.x * 128;
  const int wr   = (wid & 3) * 32;
  const int wc   = (wid >> 2) * 64;

  v8f acc[2][4];
#pragma unroll
  for (int a = 0; a < 2; ++a)
#pragma unroll
    for (int b = 0; b < 4; ++b)
#pragma unroll
      for (int i = 0; i < 8; ++i) acc[a][b][i] = 0.0f;

  for (int k0 = 0; k0 < DIM; k0 += 32) {
    __syncthreads();
    load_tile_f32(As, X, m0, DIM, k0, tid);
    load_tile_f32(Bs, W, n0, DIM, k0, tid);
    __syncthreads();
    v16h a0 = frag_a_h<LDH>(As, wr, 0, lane);
    v16h a1 = frag_a_h<LDH>(As, wr + 16, 0, lane);
#pragma unroll
    for (int j = 0; j < 4; ++j) {
      v16h b = frag_b_h<LDH>(Bs, wc + j * 16, 0, lane);
      acc[0][j] = wmma_f16(a0, b, acc[0][j]);
      acc[1][j] = wmma_f16(a1, b, acc[1][j]);
    }
  }

  const int h2 = lane >> 4;
#pragma unroll
  for (int tr = 0; tr < 2; ++tr) {
#pragma unroll
    for (int tc = 0; tc < 4; ++tc) {
      const int n  = n0 + wc + tc * 16 + (lane & 15);
      const float bv = bias[n];
#pragma unroll
      for (int i = 0; i < 8; ++i) {
        const int m = m0 + wr + tr * 16 + h2 * 8 + i;
        const float v = acc[tr][tc][i] + bv;
        if constexpr (MODE == 0) {
          const int bb = m >> 10, s = m & (SEQ - 1);
          const int hh = n >> 6,  dk = n & (DK - 1);
          ((_Float16*)out)[(((size_t)(bb * NH + hh)) * SEQ + s) * DK + dk] =
              (_Float16)v;
        } else if constexpr (MODE == 1) {
          const int bb = m >> 10, s = m & (SEQ - 1);
          const int hh = n >> 6,  dk = n & (DK - 1);
          ((_Float16*)out)[(((size_t)(bb * NH + hh)) * DK + dk) * SEQ + s] =
              (_Float16)v;
        } else {
          ((float*)out)[(size_t)m * DIM + n] = v;
        }
      }
    }
  }
}

// ---------------- Fused attention ------------------------------------------
// One block = one (b,h) x 64-query stripe. 512 threads = 16 waves; wave w owns
// the 16x16 tile (tr = w&3, tc = w>>2) of every 64x64 working set.
// Dynamic LDS: scores f32 [64][SC_LD] + Q/K/V f16 tiles [64][TLD].
__global__ void __launch_bounds__(512)
fused_attn(const _Float16* __restrict__ qh, const _Float16* __restrict__ kh,
           const _Float16* __restrict__ vT, float* __restrict__ att,
           float* __restrict__ ctx) {
  extern __shared__ char smem_raw[];
  float*    Ssc = (float*)smem_raw;
  _Float16* Qs  = (_Float16*)(smem_raw + (size_t)64 * SC_LD * 4);
  _Float16* Ks  = Qs + 64 * TLD;
  _Float16* Vs  = Ks + 64 * TLD;

  const int tid  = threadIdx.x;   // 0..511
  const int lane = tid & 31;
  const int wid  = tid >> 5;      // 0..15
  const int tr   = wid & 3;       // tile row (16 rows)
  const int tc   = wid >> 2;      // tile col (16 cols)
  const int h2   = lane >> 4;

  const int bh = blockIdx.y;
  const int b  = bh >> 4;
  const int h  = bh & (NH - 1);
  const int m0 = blockIdx.x * 64;

  const _Float16* Q  = qh + (size_t)bh * SEQ * DK;
  const _Float16* Kp = kh + (size_t)bh * SEQ * DK;
  const _Float16* V  = vT + (size_t)bh * DK * SEQ;

  const int ntiles = (m0 >> 6) + 1;      // causal: key tiles with kb <= m0
  const int kLimit = m0 + 64;

  // ---- load Q stripe once: 64 x 64 halves, 8 halves/thread ----
  {
    const int r = tid >> 3, c = (tid & 7) * 8;
    *(v8h*)(Qs + r * TLD + c) = *(const v8h*)(Q + (size_t)(m0 + r) * DK + c);
  }

  // ---- Phase 1: scores = Q K^T * 0.125 (+ causal -1e9) -> LDS ----
  for (int kt = 0; kt < ntiles; ++kt) {
    const int kb = kt * 64;
    __syncthreads();
    {
      const int r = tid >> 3, c = (tid & 7) * 8;
      *(v8h*)(Ks + r * TLD + c) = *(const v8h*)(Kp + (size_t)(kb + r) * DK + c);
    }
    __syncthreads();

    v8f acc;
#pragma unroll
    for (int i = 0; i < 8; ++i) acc[i] = 0.0f;
#pragma unroll
    for (int k0 = 0; k0 < DK; k0 += 32) {
      v16h a = frag_a_h<TLD>(Qs, tr * 16, k0, lane);
      v16h bf = frag_b_h<TLD>(Ks, tc * 16, k0, lane);
      acc = wmma_f16(a, bf, acc);
    }

    const int n_loc = tc * 16 + (lane & 15);
    const int n     = kb + n_loc;
#pragma unroll
    for (int i = 0; i < 8; ++i) {
      const int m_loc = tr * 16 + h2 * 8 + i;
      const int m     = m0 + m_loc;
      float v = acc[i] * 0.125f;
      if (n > m) v += -1e9f;
      Ssc[(size_t)m_loc * SC_LD + n] = v;
    }
  }
  __syncthreads();

  // ---- Phase 2: row softmax in LDS; write att (once) to HBM ----
  float* attBH = att + (size_t)bh * SEQ * SEQ;
#pragma unroll
  for (int rr = 0; rr < 4; ++rr) {
    const int row = wid * 4 + rr;
    const int m   = m0 + row;
    float* srow = Ssc + (size_t)row * SC_LD;

    float mx = -3.0e38f;
    for (int c = lane; c < kLimit; c += 32) mx = fmaxf(mx, srow[c]);
#pragma unroll
    for (int d = 16; d > 0; d >>= 1) mx = fmaxf(mx, __shfl_xor(mx, d, 32));

    float sum = 0.0f;
    for (int c = lane; c < kLimit; c += 32) {
      const float e = __expf(srow[c] - mx);
      srow[c] = e;
      sum += e;
    }
#pragma unroll
    for (int d = 16; d > 0; d >>= 1) sum += __shfl_xor(sum, d, 32);
    const float inv = 1.0f / sum;

    float* arow = attBH + (size_t)m * SEQ;
    for (int c = lane; c < kLimit; c += 32) {
      const float p = srow[c] * inv;
      srow[c] = p;
      arow[c] = p;
    }
    for (int c = kLimit + lane; c < SEQ; c += 32) arow[c] = 0.0f;
  }
  __syncthreads();

  // ---- Phase 3: ctx = P @ V (A from LDS f32->f16, B = pre-transposed V) ----
  v8f acc;
#pragma unroll
  for (int i = 0; i < 8; ++i) acc[i] = 0.0f;

  for (int kt = 0; kt < ntiles; ++kt) {
    const int kb = kt * 64;
    __syncthreads();
    {
      const int r = tid >> 3, c = (tid & 7) * 8;
      *(v8h*)(Vs + r * TLD + c) = *(const v8h*)(V + (size_t)r * SEQ + kb + c);
    }
    __syncthreads();
#pragma unroll
    for (int k0 = 0; k0 < 64; k0 += 32) {
      v16h a  = frag_a_f32(Ssc, tr * 16, kb + k0, lane);
      v16h bf = frag_b_h<TLD>(Vs, tc * 16, k0, lane);
      acc = wmma_f16(a, bf, acc);
    }
  }

  const int dk = tc * 16 + (lane & 15);
#pragma unroll
  for (int i = 0; i < 8; ++i) {
    const int m = m0 + tr * 16 + h2 * 8 + i;
    ctx[((size_t)b * SEQ + m) * DIM + h * DK + dk] = acc[i];
  }
}

// ---------------------------------------------------------------------------

extern "C" void kernel_launch(void* const* d_in, const int* in_sizes, int n_in,
                              void* d_out, int out_size, void* d_ws,
                              size_t ws_size, hipStream_t stream) {
  (void)in_sizes; (void)n_in; (void)out_size; (void)ws_size;

  const float* query = (const float*)d_in[0];
  const float* key   = (const float*)d_in[1];
  const float* value = (const float*)d_in[2];
  // d_in[3] is the causal mask; causality is applied analytically.
  const float* Wq = (const float*)d_in[4];
  const float* bq = (const float*)d_in[5];
  const float* Wk = (const float*)d_in[6];
  const float* bk = (const float*)d_in[7];
  const float* Wv = (const float*)d_in[8];
  const float* bv = (const float*)d_in[9];
  const float* Wo = (const float*)d_in[10];
  const float* bo = (const float*)d_in[11];

  const size_t headElems = (size_t)BATCH * NH * SEQ * DK;  // 4,194,304
  _Float16* qh  = (_Float16*)d_ws;
  _Float16* kh  = qh + headElems;
  _Float16* vT  = kh + headElems;
  float*    ctx = (float*)(vT + headElems);

  float* att = (float*)d_out;
  float* out = att + (size_t)BATCH * NH * SEQ * SEQ;

  const dim3 blk(256);
  proj_kernel<0><<<dim3(8, 32), blk, 0, stream>>>(query, Wq, bq, (void*)qh);
  proj_kernel<0><<<dim3(8, 32), blk, 0, stream>>>(key,   Wk, bk, (void*)kh);
  proj_kernel<1><<<dim3(8, 32), blk, 0, stream>>>(value, Wv, bv, (void*)vT);

  // Fused scores+softmax+P@V: LDS = 64*SC_LD*4 (scores) + 3 * 64*TLD*2 (tiles)
  const size_t smem = (size_t)64 * SC_LD * 4 + 3 * (size_t)64 * TLD * 2;
  fused_attn<<<dim3(SEQ / 64, BATCH * NH), dim3(512), smem, stream>>>(
      qh, kh, vT, att, ctx);

  proj_kernel<2><<<dim3(8, 32), blk, 0, stream>>>(ctx, Wo, bo, (void*)out);
}
